// FeatureCorrelation_16003048145422
// MI455X (gfx1250) — matrix operand order repaired and checked by reading, not compiled, not run on hardware
//
#include <hip/hip_runtime.h>
#include <hip/hip_bf16.h>
#include <math.h>

#define NB   8
#define NC   256
#define HW   1600
#define NP1  1601
#define ZLD  1604   // padded row stride (16B aligned)

typedef __bf16 bf16;
typedef bf16  v16bf __attribute__((ext_vector_type(16)));
typedef bf16  v8bf  __attribute__((ext_vector_type(8)));
typedef float v8f   __attribute__((ext_vector_type(8)));

// ---------------------------------------------------------------------------
// Pack f32 features -> bf16, K-contiguous row-major [pos][c] layouts.
// faT[b][m][c] = A[b][c][m%40][m/40]   (w-major flatten of A)
// fbT[b][n][c] = B[b][c][n/40][n%40]
// ---------------------------------------------------------------------------
__global__ void k_pack(const float* __restrict__ A, const float* __restrict__ Bf,
                       bf16* __restrict__ faT, bf16* __restrict__ fbT) {
    int p = blockIdx.x;              // b*HW + pos
    int c = threadIdx.x;             // 0..255
    int b = p / HW;
    int m = p - b * HW;
    int ha = m % 40, wa = m / 40;
    size_t ai = ((size_t)(b * NC + c)) * HW + ha * 40 + wa;
    size_t bi = ((size_t)(b * NC + c)) * HW + m;
    faT[(size_t)p * NC + c] = (bf16)A[ai];
    fbT[(size_t)p * NC + c] = (bf16)Bf[bi];
}

// ---------------------------------------------------------------------------
// corr = faT^T * fbT  per batch (M=N=1600, K=256), relu'd, written into the
// interior of Z.  64x64 tile per workgroup, 8 waves, 2 WMMA tiles per wave,
// v_wmma_f32_16x16x32_bf16, K unrolled (8 steps).
// ---------------------------------------------------------------------------
__global__ void __launch_bounds__(256)
k_gemm(const bf16* __restrict__ faT, const bf16* __restrict__ fbT,
       float* __restrict__ Z) {
    const int bb = blockIdx.z, tm = blockIdx.y, tn = blockIdx.x;
    const int wave = threadIdx.x >> 5, lane = threadIdx.x & 31;
    const int mi = wave & 3;                 // 4 M-subtiles
    const int nj = (wave >> 2) << 1;         // 2 waves cover 4 N-subtiles (2 each)
    const int laneR = lane & 15, hi = lane >> 4;

    const int m_base  = tm * 64 + mi * 16;
    const int n_base0 = tn * 64 + nj * 16;
    const int n_base1 = n_base0 + 16;

    const bf16* rowA  = faT + ((size_t)bb * HW + m_base  + laneR) * NC;
    const bf16* rowB0 = fbT + ((size_t)bb * HW + n_base0 + laneR) * NC;
    const bf16* rowB1 = fbT + ((size_t)bb * HW + n_base1 + laneR) * NC;

    v8f acc0 = {}; v8f acc1 = {};
#pragma unroll
    for (int k0 = 0; k0 < NC; k0 += 32) {
        union { v16bf v; v8bf h[2]; } a, b0, b1;
        // A, contiguous k order chosen here (matches B): lanes<16 hold K 0..15; lanes>=16 hold K 16..31
        a.h[0]  = *(const v8bf*)(rowA  + k0 + hi * 16);
        a.h[1]  = *(const v8bf*)(rowA  + k0 + hi * 16 + 8);
        // B, contiguous k order chosen here (matches A): lanes<16 hold K 0..15; lanes>=16 hold K 16..31
        b0.h[0] = *(const v8bf*)(rowB0 + k0 + hi * 16);
        b0.h[1] = *(const v8bf*)(rowB0 + k0 + hi * 16 + 8);
        b1.h[0] = *(const v8bf*)(rowB1 + k0 + hi * 16);
        b1.h[1] = *(const v8bf*)(rowB1 + k0 + hi * 16 + 8);
        acc0 = __builtin_amdgcn_wmma_f32_16x16x32_bf16(false, a.v, false, b0.v,
                                                       (short)0, acc0, false, false);
        acc1 = __builtin_amdgcn_wmma_f32_16x16x32_bf16(false, a.v, false, b1.v,
                                                       (short)0, acc1, false, false);
    }
    float* zb = Z + (size_t)bb * NP1 * ZLD;
#pragma unroll
    for (int i = 0; i < 8; ++i) {
        int row = m_base + i + 8 * hi;       // C/D layout: VGPR i -> M=i (+8 for hi lanes)
        zb[(size_t)row * ZLD + n_base0 + laneR] = fmaxf(acc0[i], 0.f);
        zb[(size_t)row * ZLD + n_base1 + laneR] = fmaxf(acc1[i], 0.f);
    }
}

// ---------------------------------------------------------------------------
// Column L2 norm (over m) of relu'd corr; fold in the 1/sqrt(C)=1/16 scale.
// ---------------------------------------------------------------------------
__global__ void k_colnorm(const float* __restrict__ Z, float* __restrict__ inv) {
    int n = blockIdx.x * 256 + threadIdx.x;
    int b = blockIdx.y;
    if (n >= HW) return;
    const float* zb = Z + (size_t)b * NP1 * ZLD;
    float s = 0.f;
    for (int m = 0; m < HW; ++m) { float v = zb[(size_t)m * ZLD + n]; s += v * v; }
    inv[b * HW + n] = 1.f / (sqrtf(s + 1e-6f) * 16.f);
}

// ---------------------------------------------------------------------------
// Scale interior by inv[n]; fill bin row/col/corner with alpha.
// ---------------------------------------------------------------------------
__global__ void k_scale_border(float* __restrict__ Z, const float* __restrict__ inv,
                               const float* __restrict__ alphap) {
    int idx = blockIdx.x * 256 + threadIdx.x;
    int b = blockIdx.y;
    if (idx >= NP1 * NP1) return;
    int m = idx / NP1, n = idx - m * NP1;
    float* zb = Z + (size_t)b * NP1 * ZLD;
    if (m < HW && n < HW) zb[(size_t)m * ZLD + n] *= inv[b * HW + n];
    else                  zb[(size_t)m * ZLD + n]  = *alphap;
}

__global__ void k_init_uv(float* __restrict__ u, float* __restrict__ v) {
    int i = blockIdx.x * 256 + threadIdx.x;
    if (i < NB * NP1) { u[i] = 0.f; v[i] = 0.f; }
}

// ---------------------------------------------------------------------------
// u[m] = log_mu[m] - logsumexp_n( Z[m,n] + v[n] )  — one workgroup per (b,m),
// fully coalesced row stream, online (single-pass) logsumexp.
// ---------------------------------------------------------------------------
__global__ void __launch_bounds__(256)
k_row_lse(const float* __restrict__ Z, const float* __restrict__ v,
          float* __restrict__ u, float norm_c, float logns) {
    int m = blockIdx.x, b = blockIdx.y;
    const float* zb = Z + (size_t)b * NP1 * ZLD + (size_t)m * ZLD;
    const float* vb = v + b * NP1;
    float mx = -INFINITY, sm = 0.f;
    for (int n = threadIdx.x; n < NP1; n += 256) {
        float x = zb[n] + vb[n];
        if (x > mx) { sm = sm * __expf(mx - x) + 1.f; mx = x; }
        else        { sm += __expf(x - mx); }
    }
    __shared__ float smx[256], ssm[256];
    smx[threadIdx.x] = mx; ssm[threadIdx.x] = sm;
    __syncthreads();
    for (int s = 128; s > 0; s >>= 1) {
        if (threadIdx.x < (unsigned)s) {
            float m1 = smx[threadIdx.x],     s1 = ssm[threadIdx.x];
            float m2 = smx[threadIdx.x + s], s2 = ssm[threadIdx.x + s];
            float M = fmaxf(m1, m2);
            smx[threadIdx.x] = M;
            ssm[threadIdx.x] = s1 * __expf(m1 - M) + s2 * __expf(m2 - M);
        }
        __syncthreads();
    }
    if (threadIdx.x == 0) {
        float lse = smx[0] + __logf(ssm[0]);
        float lmu = (m < HW) ? norm_c : (logns + norm_c);
        u[b * NP1 + m] = lmu - lse;
    }
}

// ---------------------------------------------------------------------------
// v[n] = log_nu[n] - logsumexp_m( Z[m,n] + u[m] )  — 64 columns per workgroup,
// 4 row-phases per column so each 64-thread group reads 256B-contiguous spans.
// ---------------------------------------------------------------------------
__global__ void __launch_bounds__(256)
k_col_lse(const float* __restrict__ Z, const float* __restrict__ u,
          float* __restrict__ v, float norm_c, float logns) {
    int col  = blockIdx.x * 64 + (threadIdx.x & 63);
    int quad = threadIdx.x >> 6;
    int b = blockIdx.y;
    const float* zb = Z + (size_t)b * NP1 * ZLD;
    const float* ub = u + b * NP1;
    float mx = -INFINITY, sm = 0.f;
    if (col < NP1) {
        for (int m = quad; m < NP1; m += 4) {
            float x = zb[(size_t)m * ZLD + col] + ub[m];
            if (x > mx) { sm = sm * __expf(mx - x) + 1.f; mx = x; }
            else        { sm += __expf(x - mx); }
        }
    }
    __shared__ float smx[256], ssm[256];
    smx[threadIdx.x] = mx; ssm[threadIdx.x] = sm;
    __syncthreads();
    if (quad == 0 && col < NP1) {
        float M = mx, S = sm;
#pragma unroll
        for (int q = 1; q < 4; ++q) {
            float m2 = smx[threadIdx.x + 64 * q], s2 = ssm[threadIdx.x + 64 * q];
            float Mn = fmaxf(M, m2);
            S = S * __expf(M - Mn) + s2 * __expf(m2 - Mn);
            M = Mn;
        }
        float lse = M + __logf(S);
        float lnu = (col < HW) ? norm_c : (logns + norm_c);
        v[b * NP1 + col] = lnu - lse;
    }
}

// ---------------------------------------------------------------------------
// Per-row partial of  sum exp(Z+u+v-norm)*Z  over interior (deterministic).
// ---------------------------------------------------------------------------
__global__ void __launch_bounds__(256)
k_final(const float* __restrict__ Z, const float* __restrict__ u,
        const float* __restrict__ v, float* __restrict__ part, float norm_c) {
    int m = blockIdx.x, b = blockIdx.y;
    const float* zb = Z + (size_t)b * NP1 * ZLD + (size_t)m * ZLD;
    const float* vb = v + b * NP1;
    float um = u[b * NP1 + m];
    float s = 0.f;
    for (int n = threadIdx.x; n < HW; n += 256) {
        float zv = zb[n];
        s += __expf(zv + um + vb[n] - norm_c) * zv;
    }
    __shared__ float red[256];
    red[threadIdx.x] = s; __syncthreads();
    for (int t = 128; t > 0; t >>= 1) {
        if (threadIdx.x < (unsigned)t) red[threadIdx.x] += red[threadIdx.x + t];
        __syncthreads();
    }
    if (threadIdx.x == 0) part[b * HW + m] = red[0];
}

__global__ void k_out(const float* __restrict__ part, float* __restrict__ out) {
    int b = blockIdx.x;
    float s = 0.f;
    for (int m = threadIdx.x; m < HW; m += 256) s += part[b * HW + m];
    __shared__ float red[256];
    red[threadIdx.x] = s; __syncthreads();
    for (int t = 128; t > 0; t >>= 1) {
        if (threadIdx.x < (unsigned)t) red[threadIdx.x] += red[threadIdx.x + t];
        __syncthreads();
    }
    if (threadIdx.x == 0) out[b] = __expf(-red[0]);
}

// ---------------------------------------------------------------------------
extern "C" void kernel_launch(void* const* d_in, const int* in_sizes, int n_in,
                              void* d_out, int out_size, void* d_ws, size_t ws_size,
                              hipStream_t stream) {
    const float* A     = (const float*)d_in[0];
    const float* Bf    = (const float*)d_in[1];
    const float* alpha = (const float*)d_in[2];
    float* out = (float*)d_out;

    char* ws = (char*)d_ws;
    size_t off = 0;
    bf16*  faT = (bf16*)(ws + off);  off += (size_t)NB * HW * NC * sizeof(bf16);
    bf16*  fbT = (bf16*)(ws + off);  off += (size_t)NB * HW * NC * sizeof(bf16);
    float* Z   = (float*)(ws + off); off += (size_t)NB * NP1 * ZLD * sizeof(float);
    float* u   = (float*)(ws + off); off += (size_t)NB * NP1 * sizeof(float);
    float* v   = (float*)(ws + off); off += (size_t)NB * NP1 * sizeof(float);
    float* inv = (float*)(ws + off); off += (size_t)NB * HW * sizeof(float);
    float* part = inv;  // reuse inv buffer after scale_border for final partials

    const float norm_c = -logf((float)(HW + HW));  // -log(3200)
    const float logns  = logf((float)HW);          //  log(1600)

    k_init_uv<<<dim3((NB * NP1 + 255) / 256), 256, 0, stream>>>(u, v);
    k_pack<<<dim3(NB * HW), 256, 0, stream>>>(A, Bf, faT, fbT);
    k_gemm<<<dim3(HW / 64, HW / 64, NB), 256, 0, stream>>>(faT, fbT, Z);
    k_colnorm<<<dim3((HW + 255) / 256, NB), 256, 0, stream>>>(Z, inv);
    k_scale_border<<<dim3((NP1 * NP1 + 255) / 256, NB), 256, 0, stream>>>(Z, inv, alpha);

    for (int it = 0; it < 100; ++it) {
        k_row_lse<<<dim3(NP1, NB), 256, 0, stream>>>(Z, v, u, norm_c, logns);
        k_col_lse<<<dim3((NP1 + 63) / 64, NB), 256, 0, stream>>>(Z, u, v, norm_c, logns);
    }

    k_final<<<dim3(HW, NB), 256, 0, stream>>>(Z, u, v, part, norm_c);
    k_out<<<dim3(NB), 256, 0, stream>>>(part, out);
}
